// Atten_38826504356084
// MI455X (gfx1250) — matrix operand order repaired and checked by reading, not compiled, not run on hardware
//
#include <hip/hip_runtime.h>

typedef __attribute__((ext_vector_type(16))) _Float16 v16h;
typedef __attribute__((ext_vector_type(8)))  _Float16 v8h;
typedef __attribute__((ext_vector_type(8)))  float    v8f;

union V16U { v16h v; v8h h[2]; };

// A fragment: M x 32 tile, row-major in memory with leading dim `ld` (K contiguous).
// Lane L: row = L%16, K = (L/16)*8 + {0..7}  and  16 + (L/16)*8 + {0..7}.
static __device__ __forceinline__ v16h frag_a(const _Float16* base, int ld, int lane) {
  const int l = lane & 15, g = lane >> 4;
  const _Float16* p = base + l * ld + g * 8;
  V16U u;
  u.h[0] = *(const v8h*)(p);
  u.h[1] = *(const v8h*)(p + 16);
  return u.v;
}

// B fragment: 32 x 16 tile given as B^T row-major (rows indexed by output col,
// K contiguous with leading dim `ld`). Lane L: col = L%16, K = (L/16)*8 + {0..7}  and  16 + (L/16)*8 + {0..7}.
static __device__ __forceinline__ v16h frag_b(const _Float16* base, int ld, int lane) {
  const int l = lane & 15, g = lane >> 4;
  const _Float16* p = base + l * ld + g * 8;
  V16U u;
  u.h[0] = *(const v8h*)(p);
  u.h[1] = *(const v8h*)(p + 16);
  return u.v;
}

static __device__ __forceinline__ v8f wmma_f16(v16h a, v16h b, v8f c) {
  return __builtin_amdgcn_wmma_f32_16x16x32_f16(false, a, false, b, (short)0, c,
                                                false, false);
}

// 16B async copy global -> LDS, tracked by ASYNCcnt (CDNA5 ISA).
// LDS byte address = low 32 bits of the generic pointer (ISA 10.2 aperture rule).
static __device__ __forceinline__ void async_cp16(_Float16* lds_dst,
                                                  const _Float16* gsrc) {
  unsigned int ldsa = (unsigned int)(unsigned long long)lds_dst;
  asm volatile("global_load_async_to_lds_b128 %0, %1, off"
               :: "v"(ldsa), "v"((unsigned long long)gsrc)
               : "memory");
}

static __device__ __forceinline__ void wait_async0() {
  asm volatile("s_wait_asynccnt 0x0" ::: "memory");
}

// ---------------------------------------------------------------- convert f32 -> f16
__global__ void cvt_f32_f16(const float* __restrict__ in, _Float16* __restrict__ out,
                            int n) {
  int i = blockIdx.x * blockDim.x + threadIdx.x;
  if (i < n) out[i] = (_Float16)in[i];
}

// ---------------------------------------------------------------- QKV projection GEMM
// C[4096][1536] = Xh[4096][512] @ Wqkv^T + bqkv ; scatter into Q/K/Vt (f16).
__global__ void __launch_bounds__(128)
qkv_gemm(const _Float16* __restrict__ X, const _Float16* __restrict__ W,
         const float* __restrict__ bias, const float* __restrict__ scale_p,
         _Float16* __restrict__ Qh, _Float16* __restrict__ Kh,
         _Float16* __restrict__ Vth) {
  const int lane = threadIdx.x & 31;
  const int wv = threadIdx.x >> 5;
  const int m0 = blockIdx.y * 64 + (wv >> 1) * 32;
  const int n0 = blockIdx.x * 64 + (wv & 1) * 32;

  v8f acc[2][2] = {};
  for (int k0 = 0; k0 < 512; k0 += 32) {
    v16h a0 = frag_a(X + (size_t)m0 * 512 + k0, 512, lane);
    v16h a1 = frag_a(X + (size_t)(m0 + 16) * 512 + k0, 512, lane);
    v16h b0 = frag_b(W + (size_t)n0 * 512 + k0, 512, lane);
    v16h b1 = frag_b(W + (size_t)(n0 + 16) * 512 + k0, 512, lane);
    acc[0][0] = wmma_f16(a0, b0, acc[0][0]);
    acc[0][1] = wmma_f16(a0, b1, acc[0][1]);
    acc[1][0] = wmma_f16(a1, b0, acc[1][0]);
    acc[1][1] = wmma_f16(a1, b1, acc[1][1]);
  }

  const float sc = scale_p[0];
  const int l = lane & 15, g = lane >> 4;
  for (int j = 0; j < 2; ++j) {
    const int col = n0 + j * 16 + l;       // 0..1535
    const int which = col >> 9;            // 0=Q 1=K 2=V
    const int hh = (col & 511) >> 6;       // head
    const int dd = col & 63;               // within-head dim
    const float bv = bias[col];
    for (int i = 0; i < 2; ++i) {
#pragma unroll
      for (int r = 0; r < 8; ++r) {
        const int m = m0 + i * 16 + g * 8 + r;
        const float v = acc[i][j][r] + bv;
        if (which == 0)
          Qh[((size_t)hh * 4096 + m) * 64 + dd] = (_Float16)(v * sc);
        else if (which == 1)
          Kh[((size_t)hh * 4096 + m) * 64 + dd] = (_Float16)v;
        else
          Vth[((size_t)hh * 64 + dd) * 4096 + m] = (_Float16)v;
      }
    }
  }
}

// ---------------------------------------------------------------- flash attention
// grid (64, 8), block 128 (4 waves); wave owns 16 query rows; 64 keys per step.
// Double-buffered K/V tiles staged with async-to-LDS; one barrier per step.
__global__ void __launch_bounds__(128)
attn_kernel(const _Float16* __restrict__ Qh, const _Float16* __restrict__ Kh,
            const _Float16* __restrict__ Vth, _Float16* __restrict__ Oh) {
  __shared__ alignas(16) _Float16 sKb[2][64 * 64];  // [buf][key][d]   8KB each
  __shared__ alignas(16) _Float16 sVb[2][64 * 64];  // [buf][d][key]   8KB each
  __shared__ alignas(16) _Float16 sP[4][16 * 64];   // per-wave P tile 2KB each

  const int tid = threadIdx.x;
  const int lane = tid & 31, wv = tid >> 5;
  const int l = lane & 15, g = lane >> 4;
  const int h = blockIdx.y;
  const int q0 = blockIdx.x * 64 + wv * 16;

  // Q fragments (row-major, d contiguous), resident for the whole loop.
  const _Float16* Qbase = Qh + ((size_t)h * 4096 + q0) * 64;
  const v16h qa0 = frag_a(Qbase + 0, 64, lane);
  const v16h qa1 = frag_a(Qbase + 32, 64, lane);

  v8f acc[4] = {};
  float mrow[8], lrow[8];
#pragma unroll
  for (int r = 0; r < 8; ++r) { mrow[r] = -1e30f; lrow[r] = 0.f; }

  // ---- async staging of one 64-key tile (K: 8KB contiguous, Vt: 64 rows x 128B)
  auto stage = [&](int buf, int j0) {
    const _Float16* Kg = Kh + ((size_t)h * 4096 + j0) * 64;
    _Float16* dK = &sKb[buf][0];
#pragma unroll
    for (int i = 0; i < 4; ++i) {
      const int c = tid + i * 128;                 // 512 x 16B chunks
      async_cp16(dK + c * 8, Kg + c * 8);
    }
    _Float16* dV = &sVb[buf][0];
#pragma unroll
    for (int i = 0; i < 4; ++i) {
      const int c = tid + i * 128;
      const int dd = c >> 3, off = (c & 7) * 8;    // row d, 8 chunks per row
      async_cp16(dV + dd * 64 + off,
                 Vth + ((size_t)h * 64 + dd) * 4096 + j0 + off);
    }
  };

  stage(0, 0);

  for (int j0 = 0; j0 < 4096; j0 += 64) {
    const int cur = (j0 >> 6) & 1;
    wait_async0();        // own async copies done...
    __syncthreads();      // ...and everyone's; also orders last step's LDS reads
    if (j0 + 64 < 4096) {
      stage(cur ^ 1, j0 + 64);          // overlaps with compute below
      if (j0 + 128 < 4096) {            // warm GL2 two steps ahead
        if (tid < 64)
          __builtin_prefetch(Kh + ((size_t)h * 4096 + j0 + 128) * 64 + tid * 64, 0, 1);
        else
          __builtin_prefetch(Vth + ((size_t)h * 64 + (tid - 64)) * 4096 + j0 + 128, 0, 1);
      }
    }

    // S = Q K^T : four 16-key tiles, d=64 -> 2 WMMAs each.
    const _Float16* bK = &sKb[cur][0];
    v8f s[4];
#pragma unroll
    for (int nt = 0; nt < 4; ++nt) {
      v8f c0 = {};
      v16h bk0 = frag_b(bK + nt * 16 * 64, 64, lane);
      v16h bk1 = frag_b(bK + nt * 16 * 64 + 32, 64, lane);
      c0 = wmma_f16(qa0, bk0, c0);
      c0 = wmma_f16(qa1, bk1, c0);
      s[nt] = c0;
    }

    // Online softmax. D-fragment: lane holds col n, rows (g*8+r); each row lives
    // in one 16-lane half, so xor-shuffles 1/2/4/8 reduce exactly over the row.
    _Float16* sPw = &sP[wv][0];
#pragma unroll
    for (int r = 0; r < 8; ++r) {
      float t = fmaxf(fmaxf(s[0][r], s[1][r]), fmaxf(s[2][r], s[3][r]));
      t = fmaxf(t, __shfl_xor(t, 1, 32));
      t = fmaxf(t, __shfl_xor(t, 2, 32));
      t = fmaxf(t, __shfl_xor(t, 4, 32));
      t = fmaxf(t, __shfl_xor(t, 8, 32));
      const float mn = fmaxf(mrow[r], t);
      const float corr = __expf(mrow[r] - mn);
      mrow[r] = mn;
      const float p0 = __expf(s[0][r] - mn);
      const float p1 = __expf(s[1][r] - mn);
      const float p2 = __expf(s[2][r] - mn);
      const float p3 = __expf(s[3][r] - mn);
      float ps = (p0 + p1) + (p2 + p3);
      ps += __shfl_xor(ps, 1, 32);
      ps += __shfl_xor(ps, 2, 32);
      ps += __shfl_xor(ps, 4, 32);
      ps += __shfl_xor(ps, 8, 32);
      lrow[r] = lrow[r] * corr + ps;
#pragma unroll
      for (int dt = 0; dt < 4; ++dt) acc[dt][r] *= corr;
      const int row = (g * 8 + r) * 64;
      sPw[row + l]      = (_Float16)p0;
      sPw[row + 16 + l] = (_Float16)p1;
      sPw[row + 32 + l] = (_Float16)p2;
      sPw[row + 48 + l] = (_Float16)p3;
    }

    // O += P V : P(16x64) from LDS as two A-fragments; V from sVt (n contiguous).
    const _Float16* bV = &sVb[cur][0];
    const v16h pa0 = frag_a(sPw, 64, lane);
    const v16h pa1 = frag_a(sPw + 32, 64, lane);
#pragma unroll
    for (int dt = 0; dt < 4; ++dt) {
      v16h bv0 = frag_b(bV + dt * 16 * 64, 64, lane);
      v16h bv1 = frag_b(bV + dt * 16 * 64 + 32, 64, lane);
      acc[dt] = wmma_f16(pa0, bv0, acc[dt]);
      acc[dt] = wmma_f16(pa1, bv1, acc[dt]);
    }
  }

  // Normalize rows and store f16 attention output [4096][512].
#pragma unroll
  for (int dt = 0; dt < 4; ++dt) {
#pragma unroll
    for (int r = 0; r < 8; ++r) {
      const float o = acc[dt][r] / lrow[r];
      Oh[(size_t)(q0 + g * 8 + r) * 512 + h * 64 + dt * 16 + l] = (_Float16)o;
    }
  }
}

// ---------------------------------------------------------------- output projection
__global__ void __launch_bounds__(128)
proj_gemm(const _Float16* __restrict__ X, const _Float16* __restrict__ W,
          const float* __restrict__ bias, float* __restrict__ out) {
  const int lane = threadIdx.x & 31;
  const int wv = threadIdx.x >> 5;
  const int m0 = blockIdx.y * 64 + (wv >> 1) * 32;
  const int n0 = blockIdx.x * 64 + (wv & 1) * 32;

  v8f acc[2][2] = {};
  for (int k0 = 0; k0 < 512; k0 += 32) {
    v16h a0 = frag_a(X + (size_t)m0 * 512 + k0, 512, lane);
    v16h a1 = frag_a(X + (size_t)(m0 + 16) * 512 + k0, 512, lane);
    v16h b0 = frag_b(W + (size_t)n0 * 512 + k0, 512, lane);
    v16h b1 = frag_b(W + (size_t)(n0 + 16) * 512 + k0, 512, lane);
    acc[0][0] = wmma_f16(a0, b0, acc[0][0]);
    acc[0][1] = wmma_f16(a0, b1, acc[0][1]);
    acc[1][0] = wmma_f16(a1, b0, acc[1][0]);
    acc[1][1] = wmma_f16(a1, b1, acc[1][1]);
  }

  const int l = lane & 15, g = lane >> 4;
  for (int j = 0; j < 2; ++j) {
    const int col = n0 + j * 16 + l;
    const float bv = bias[col];
    for (int i = 0; i < 2; ++i) {
#pragma unroll
      for (int r = 0; r < 8; ++r) {
        const int m = m0 + i * 16 + g * 8 + r;
        out[(size_t)m * 512 + col] = acc[i][j][r] + bv;
      }
    }
  }
}

// ---------------------------------------------------------------- launch
extern "C" void kernel_launch(void* const* d_in, const int* in_sizes, int n_in,
                              void* d_out, int out_size, void* d_ws, size_t ws_size,
                              hipStream_t stream) {
  const float* x     = (const float*)d_in[0];
  const float* scale = (const float*)d_in[3];
  const float* Wqkv  = (const float*)d_in[4];
  const float* bqkv  = (const float*)d_in[5];
  const float* Wproj = (const float*)d_in[6];
  const float* bproj = (const float*)d_in[7];

  const int NX = 4096 * 512;      // 2,097,152
  const int NWQ = 1536 * 512;     //   786,432
  const int NWP = 512 * 512;      //   262,144
  const int NHD = 8 * 4096 * 64;  // 2,097,152

  _Float16* Xh     = (_Float16*)d_ws;
  _Float16* Wqkvh  = Xh + NX;
  _Float16* Wprojh = Wqkvh + NWQ;
  _Float16* Qh     = Wprojh + NWP;
  _Float16* Kh     = Qh + NHD;
  _Float16* Vth    = Kh + NHD;
  _Float16* Oh     = Vth + NHD;   // total ~23 MB of f16 workspace

  cvt_f32_f16<<<(NX + 255) / 256, 256, 0, stream>>>(x, Xh, NX);
  cvt_f32_f16<<<(NWQ + 255) / 256, 256, 0, stream>>>(Wqkv, Wqkvh, NWQ);
  cvt_f32_f16<<<(NWP + 255) / 256, 256, 0, stream>>>(Wproj, Wprojh, NWP);

  qkv_gemm<<<dim3(24, 64), 128, 0, stream>>>(Xh, Wqkvh, bqkv, scale, Qh, Kh, Vth);
  attn_kernel<<<dim3(64, 8), 128, 0, stream>>>(Qh, Kh, Vth, Oh);
  proj_gemm<<<dim3(8, 64), 128, 0, stream>>>(Oh, Wprojh, bproj, (float*)d_out);
}
